// Mamba_29996051595488
// MI455X (gfx1250) — compile-verified, matrix operand order checked by reading, not run on hardware
//
#include <hip/hip_runtime.h>

// Mamba forward, MI455X (gfx1250, wave32).
// bf16-native WMMA pipeline: activations/weights converted to bf16 once
// (weights transposed to N x K so GEMM tiles are contiguous row copies),
// then GEMM stages tiles with global_load_async_to_lds_b128 (ASYNCcnt,
// LDS double buffer) and computes with v_wmma_f32_16x16x32_bf16.
// Only out[:, -1, :] feeds the final fc, so out_proj / z / C collapse to the
// last position.

#define B_SZ     4
#define L_SZ     2048
#define NTOK     1024
#define DMODEL   1024
#define DSTATE   16
#define DCONV    4
#define DINNER   2048
#define DTRANK   64
#define DBC_P    128                     // padded dt|B|C row (96 -> 128)

typedef __attribute__((ext_vector_type(16))) __bf16 v16bf;
typedef __attribute__((ext_vector_type(8)))  __bf16 v8bf;
typedef __attribute__((ext_vector_type(8)))  float  v8f;

__device__ __forceinline__ float silu_f(float x)     { return x / (1.0f + __expf(-x)); }
__device__ __forceinline__ float softplus_f(float x) { return (x > 20.0f) ? x : log1pf(__expf(x)); }

// low 32 bits of a generic pointer to __shared__ = LDS byte offset
__device__ __forceinline__ unsigned lds_off(const void* p) {
    return (unsigned)(uintptr_t)p;
}
// async DMA: 16 bytes global -> LDS per lane, tracked by ASYNCcnt
__device__ __forceinline__ void async_b128(unsigned lds, const void* g) {
    asm volatile("global_load_async_to_lds_b128 %0, %1, off"
                 :: "v"(lds), "v"(g) : "memory");
}

// ---------------------------------------------------------------------------
// Embedding gather -> bf16: x[b,l,:] = bf16(emb[tokens[b,l],:])
// ---------------------------------------------------------------------------
__global__ void embed_kernel(const int* __restrict__ tokens,
                             const float* __restrict__ emb,
                             __bf16* __restrict__ x, int total)
{
    int idx = blockIdx.x * blockDim.x + threadIdx.x;
    if (idx >= total) return;
    int d  = idx & (DMODEL - 1);
    int bl = idx >> 10;
    int tok = tokens[bl];
    x[idx] = (__bf16)emb[(size_t)tok * DMODEL + d];
}

// ---------------------------------------------------------------------------
// Tiled transpose + convert: Wt[n][k] = bf16(W[k][n]), zero-pad n >= Nsrc.
// Grid: (K/32, Nout/32), 256 threads. Coalesced loads and stores via LDS.
// ---------------------------------------------------------------------------
#define TT 32
__global__ __launch_bounds__(256)
void transpose_cvt_kernel(const float* __restrict__ W, int ldw, int Nsrc, int K,
                          __bf16* __restrict__ Wt)
{
    __shared__ float tile[TT][TT + 1];
    const int kb = blockIdx.x * TT;
    const int nb = blockIdx.y * TT;
    const int tx = threadIdx.x & 31;
    const int ty = threadIdx.x >> 5;          // 0..7
    #pragma unroll
    for (int i = 0; i < 4; ++i) {
        int k = kb + ty + i * 8;
        int n = nb + tx;
        tile[ty + i * 8][tx] = (n < Nsrc) ? W[(size_t)k * ldw + n] : 0.0f;
    }
    __syncthreads();
    #pragma unroll
    for (int i = 0; i < 4; ++i) {
        int n = nb + ty + i * 8;
        int k = kb + tx;
        Wt[(size_t)n * K + k] = (__bf16)tile[tx][ty + i * 8];
    }
}

// ---------------------------------------------------------------------------
// WMMA GEMM: C[M,N] = epi(A[M,K] @ Wt[N,K]^T), all bf16 in, f32 acc.
// A row-major (lda), Wt row-major N x K (ldwt = K), C row-major (ldc).
// Requirements: M % 256 == 0, N % 64 == 0, K % 32 == 0 (all call sites hold).
// Block tile 256x64, 8 waves; wave w: rows 32w..32w+31 x 64 cols
//   => 2 A frags, 4 B frags, 8 WMMA per K-step.
// Staging: global_load_async_to_lds_b128 into LDS double buffer; next tile's
// 5 async ops per wave stay in flight through the WMMA block
// (s_wait_asynccnt 0x5).
// epi: 0 = none, 1 = softplus(v + bias[col]). Cbf: optional bf16 mirror of C.
// ---------------------------------------------------------------------------
#define BM 256
#define BN 64
#define BK 32

__global__ __launch_bounds__(256)
void gemm_bf16_wmma(const __bf16* __restrict__ A, int lda,
                    const __bf16* __restrict__ Wt, int ldwt,
                    float* __restrict__ C, int ldc,
                    __bf16* __restrict__ Cbf,
                    int M, int N, int K,
                    const float* __restrict__ bias, int epi)
{
    // +8 half padding => 80-byte rows: 16B aligned, bank-conflict friendly.
    __shared__ __align__(16) __bf16 As[2][BM][BK + 8];
    __shared__ __align__(16) __bf16 Bs[2][BN][BK + 8];

    const int tid  = threadIdx.x;
    const int wave = tid >> 5;
    const int lane = tid & 31;
    const int m0 = blockIdx.y * BM;
    const int n0 = blockIdx.x * BN;

    v8f acc[2][4] = {};

    const int mlane = lane & 15;
    const int kh    = (lane < 16) ? 0 : 8;   // ISA 16-bit fragment K-half select

    // A tile: 256x32 bf16 = 1024 16B chunks -> 4 per thread
    // B tile:  64x32 bf16 =  256 16B chunks -> 1 per thread
    auto stage_async = [&](int buf, int k0) {
        #pragma unroll
        for (int i = 0; i < 4; ++i) {
            int chunk = i * 256 + tid;
            int row = chunk >> 2;            // 4 chunks per 32-half row
            int c8  = (chunk & 3) << 3;      // half offset within row
            async_b128(lds_off(&As[buf][row][c8]),
                       &A[(size_t)(m0 + row) * lda + k0 + c8]);
        }
        {
            int row = tid >> 2;
            int c8  = (tid & 3) << 3;
            async_b128(lds_off(&Bs[buf][row][c8]),
                       &Wt[(size_t)(n0 + row) * ldwt + k0 + c8]);
        }
    };

    stage_async(0, 0);

    int buf = 0;
    for (int k0 = 0; k0 < K; k0 += BK) {
        const bool more = (k0 + BK) < K;
        if (more) stage_async(buf ^ 1, k0 + BK);   // next tile in flight
        if (more) asm volatile("s_wait_asynccnt 0x5" ::: "memory");
        else      asm volatile("s_wait_asynccnt 0x0" ::: "memory");
        __syncthreads();

        // A fragments: lane holds row, K = kh..kh+7 and kh+16..kh+23
        v16bf afrag[2];
        #pragma unroll
        for (int h = 0; h < 2; ++h) {
            v8bf alo = *reinterpret_cast<const v8bf*>(
                           &As[buf][wave * 32 + h * 16 + mlane][kh]);
            v8bf ahi = *reinterpret_cast<const v8bf*>(
                           &As[buf][wave * 32 + h * 16 + mlane][kh + 16]);
            #pragma unroll
            for (int i = 0; i < 8; ++i) { afrag[h][i] = alo[i]; afrag[h][i + 8] = ahi[i]; }
        }

        #pragma unroll
        for (int s = 0; s < 4; ++s) {
            v8bf blo = *reinterpret_cast<const v8bf*>(&Bs[buf][s * 16 + mlane][kh]);
            v8bf bhi = *reinterpret_cast<const v8bf*>(&Bs[buf][s * 16 + mlane][kh + 16]);
            v16bf bfrag;
            #pragma unroll
            for (int i = 0; i < 8; ++i) { bfrag[i] = blo[i]; bfrag[i + 8] = bhi[i]; }
            #pragma unroll
            for (int h = 0; h < 2; ++h) {
                acc[h][s] = __builtin_amdgcn_wmma_f32_16x16x32_bf16(
                                false, afrag[h], false, bfrag,
                                (short)0, acc[h][s], false, false);
            }
        }
        __syncthreads();
        buf ^= 1;
    }

    // C layout: VGPR r -> M = r (lanes 0-15) / r+8 (lanes 16-31), N = lane&15
    #pragma unroll
    for (int h = 0; h < 2; ++h) {
        const int rbase = m0 + wave * 32 + h * 16 + ((lane < 16) ? 0 : 8);
        #pragma unroll
        for (int s = 0; s < 4; ++s) {
            int col = n0 + s * 16 + mlane;
            #pragma unroll
            for (int r = 0; r < 8; ++r) {
                int row = rbase + r;
                float v = acc[h][s][r];
                if (epi == 1) v = softplus_f(v + bias[col]);
                C[(size_t)row * ldc + col] = v;
                if (Cbf) Cbf[(size_t)row * ldc + col] = (__bf16)v;
            }
        }
    }
}

// ---------------------------------------------------------------------------
// Depthwise causal conv (width 4) + SiLU: one thread per (b,l,d).
// Emits f32 (for scan) and bf16 (for the x_proj WMMA GEMM).
// ---------------------------------------------------------------------------
__global__ void conv_silu_kernel(const float* __restrict__ xi,
                                 const float* __restrict__ cw,
                                 const float* __restrict__ cb,
                                 float* __restrict__ xc,
                                 __bf16* __restrict__ xcb, int total)
{
    int idx = blockIdx.x * blockDim.x + threadIdx.x;
    if (idx >= total) return;
    int d  = idx & (DINNER - 1);
    int bl = idx >> 11;
    int l  = bl & (L_SZ - 1);
    float acc = cb[d];
    #pragma unroll
    for (int k = 0; k < DCONV; ++k) {
        int ls = l - (DCONV - 1) + k;
        if (ls >= 0)
            acc = fmaf(xi[idx + (size_t)(ls - l) * DINNER], cw[d * DCONV + k], acc);
    }
    float v = silu_f(acc);
    xc[idx]  = v;
    xcb[idx] = (__bf16)v;
}

// ---------------------------------------------------------------------------
// Small row-vector GEMM: out[m,n] = act(bias[n] + sum_k A[m*lda+k]*W[k*ldw+n])
// Used for M=4 projections (z_last, out_last, fc). act: 0 none, 1 silu.
// ---------------------------------------------------------------------------
template <typename TA>
__global__ void rowmat_kernel(const TA* __restrict__ A, int lda,
                              const float* __restrict__ W, int ldw,
                              const float* __restrict__ bias,
                              float* __restrict__ out, int ldo,
                              int N, int K, int act)
{
    int n = blockIdx.x * blockDim.x + threadIdx.x;
    int m = blockIdx.y;
    if (n >= N) return;
    float acc = bias ? bias[n] : 0.0f;
    const TA* arow = &A[(size_t)m * lda];
    for (int k = 0; k < K; ++k)
        acc = fmaf((float)arow[k], W[(size_t)k * ldw + n], acc);
    if (act == 1) acc = silu_f(acc);
    out[(size_t)m * ldo + n] = acc;
}

// ---------------------------------------------------------------------------
// Selective scan: one thread per (b,d) channel, h[16] in registers,
// sequential over L. B rows staged in 64-step LDS tiles (barriers /64).
// Only the last-step y survives:
//   y[b,d] = (sum_s h[s]*C_last[s] + D[d]*xc_last) * silu(z_last)
// ---------------------------------------------------------------------------
#define SCAN_TILE 64

__global__ __launch_bounds__(256)
void scan_kernel(const float* __restrict__ delta,
                 const float* __restrict__ xc,
                 const float* __restrict__ dbc,     // stride DBC_P
                 const float* __restrict__ A_log,
                 const float* __restrict__ Dv,
                 const float* __restrict__ siluz,
                 float* __restrict__ y)
{
    const int tid = threadIdx.x;
    const int b   = blockIdx.x >> 3;                 // 8 blocks per batch
    const int d   = ((blockIdx.x & 7) << 8) + tid;   // 0..2047

    float Arow[DSTATE], h[DSTATE];
    #pragma unroll
    for (int s = 0; s < DSTATE; ++s) {
        Arow[s] = -__expf(A_log[(size_t)d * DSTATE + s]);
        h[s] = 0.0f;
    }

    __shared__ float Bsh[SCAN_TILE][DSTATE];
    const size_t base = (size_t)b * L_SZ;

    for (int lb = 0; lb < L_SZ; lb += SCAN_TILE) {
        __syncthreads();
        #pragma unroll
        for (int i = 0; i < (SCAN_TILE * DSTATE) / 256; ++i) {
            int idx = i * 256 + tid;
            int ll = idx >> 4;
            int s  = idx & 15;
            Bsh[ll][s] = dbc[(base + lb + ll) * DBC_P + DTRANK + s];
        }
        __syncthreads();

        for (int l = lb; l < lb + SCAN_TILE; ++l) {
            float dt = delta[(base + l) * DINNER + d];
            float xv = xc   [(base + l) * DINNER + d];
            float dx = dt * xv;
            const float* br = Bsh[l - lb];
            #pragma unroll
            for (int s = 0; s < DSTATE; ++s)
                h[s] = fmaf(__expf(dt * Arow[s]), h[s], dx * br[s]);
        }
    }

    // final y: C at last position (uniform 16-float read, L2 broadcast)
    const float* Cl = &dbc[(base + L_SZ - 1) * DBC_P + DTRANK + DSTATE];
    float acc = 0.0f;
    #pragma unroll
    for (int s = 0; s < DSTATE; ++s) acc = fmaf(h[s], Cl[s], acc);

    float xl = xc[(base + L_SZ - 1) * DINNER + d];
    int od = b * DINNER + d;
    y[od] = fmaf(Dv[d], xl, acc) * siluz[od];
}

// ---------------------------------------------------------------------------
extern "C" void kernel_launch(void* const* d_in, const int* in_sizes, int n_in,
                              void* d_out, int out_size, void* d_ws, size_t ws_size,
                              hipStream_t stream)
{
    const int*   tokens     = (const int*)  d_in[0];
    const float* emb        = (const float*)d_in[1];
    const float* in_proj_w  = (const float*)d_in[2];
    const float* conv_w     = (const float*)d_in[3];
    const float* conv_b     = (const float*)d_in[4];
    const float* x_proj_w   = (const float*)d_in[5];
    const float* dt_proj_w  = (const float*)d_in[6];
    const float* dt_proj_b  = (const float*)d_in[7];
    const float* A_log      = (const float*)d_in[8];
    const float* Dv         = (const float*)d_in[9];
    const float* out_proj_w = (const float*)d_in[10];
    const float* fc_w       = (const float*)d_in[11];
    const float* fc_b       = (const float*)d_in[12];
    float* out = (float*)d_out;

    char* ws = (char*)d_ws;
    size_t off = 0;
    auto alloc = [&](size_t bytes) -> void* {
        void* p = ws + off;
        off = (off + bytes + 255) & ~(size_t)255;
        return p;
    };
    const int M = B_SZ * L_SZ;  // 8192 token rows

    __bf16* x_bf  = (__bf16*)alloc((size_t)M * DMODEL * 2);        // 16 MB
    __bf16* wtIn  = (__bf16*)alloc((size_t)DINNER * DMODEL * 2);   //  4 MB (2048x1024)
    __bf16* wtX   = (__bf16*)alloc((size_t)DBC_P * DINNER * 2);    //  0.5 MB (128x2048)
    __bf16* wtDt  = (__bf16*)alloc((size_t)DINNER * DTRANK * 2);   //  0.25 MB (2048x64)
    float*  xi    = (float*) alloc((size_t)M * DINNER * 4);        // 64 MB (reused as delta)
    float*  xc    = (float*) alloc((size_t)M * DINNER * 4);        // 64 MB
    __bf16* xc_bf = (__bf16*)alloc((size_t)M * DINNER * 2);        // 32 MB
    float*  dbc   = (float*) alloc((size_t)M * DBC_P * 4);         //  4 MB
    __bf16* dbc_bf= (__bf16*)alloc((size_t)M * DBC_P * 2);         //  2 MB
    float*  siluz = (float*) alloc((size_t)B_SZ * DINNER * 4);
    float*  yv    = (float*) alloc((size_t)B_SZ * DINNER * 4);
    float*  outl  = (float*) alloc((size_t)B_SZ * DMODEL * 4);
    float*  delta = xi;  // xi is dead after conv; alias for softplus(dt @ W + b)

    // 0) one-time weight transpose+convert (Wt[n][k], zero-padded rows)
    transpose_cvt_kernel<<<dim3(DMODEL / TT, DINNER / TT), 256, 0, stream>>>(
        in_proj_w, 2 * DINNER, DINNER, DMODEL, wtIn);
    transpose_cvt_kernel<<<dim3(DINNER / TT, DBC_P / TT), 256, 0, stream>>>(
        x_proj_w, DTRANK + 2 * DSTATE, DTRANK + 2 * DSTATE, DINNER, wtX);
    transpose_cvt_kernel<<<dim3(DTRANK / TT, DINNER / TT), 256, 0, stream>>>(
        dt_proj_w, DINNER, DINNER, DTRANK, wtDt);

    // 1) embedding gather -> bf16
    embed_kernel<<<dim3((M * DMODEL) / 256), 256, 0, stream>>>(tokens, emb, x_bf, M * DMODEL);

    // 2) xi = x @ in_proj_w[:, :DINNER]   (WMMA, async staging)
    gemm_bf16_wmma<<<dim3(DINNER / BN, M / BM), 256, 0, stream>>>(
        x_bf, DMODEL, wtIn, DMODEL, xi, DINNER, nullptr,
        M, DINNER, DMODEL, nullptr, 0);

    // 3) silu(z) at last position only: (4 x 2048)
    rowmat_kernel<__bf16><<<dim3(DINNER / 256, B_SZ), 256, 0, stream>>>(
        x_bf + (size_t)(L_SZ - 1) * DMODEL, L_SZ * DMODEL,
        in_proj_w + DINNER, 2 * DINNER, nullptr,
        siluz, DINNER, DINNER, DMODEL, 1);

    // 4) depthwise conv + SiLU -> xc (f32) + xc_bf (bf16)
    conv_silu_kernel<<<dim3((M * DINNER) / 256), 256, 0, stream>>>(
        xi, conv_w, conv_b, xc, xc_bf, M * DINNER);

    // 5) dbc = xc @ x_proj_w  (N padded to 128: dt|B|C)  + bf16 mirror for dt
    gemm_bf16_wmma<<<dim3(DBC_P / BN, M / BM), 256, 0, stream>>>(
        xc_bf, DINNER, wtX, DINNER, dbc, DBC_P, dbc_bf,
        M, DBC_P, DINNER, nullptr, 0);

    // 6) delta = softplus(dt @ dt_proj_w + dt_proj_b)   (WMMA, fused epilogue)
    gemm_bf16_wmma<<<dim3(DINNER / BN, M / BM), 256, 0, stream>>>(
        dbc_bf, DBC_P, wtDt, DTRANK, delta, DINNER, nullptr,
        M, DINNER, DTRANK, dt_proj_b, 1);

    // 7) selective scan -> y at last position (4 x 2048)
    scan_kernel<<<dim3(B_SZ * (DINNER / 256)), 256, 0, stream>>>(
        delta, xc, dbc, A_log, Dv, siluz, yv);

    // 8) out_last = y @ out_proj_w  (4 x 1024)
    rowmat_kernel<float><<<dim3(DMODEL / 256, B_SZ), 256, 0, stream>>>(
        yv, DINNER, out_proj_w, DMODEL, nullptr,
        outl, DMODEL, DMODEL, DINNER, 0);

    // 9) logits = out_last @ fc_w + fc_b  -> d_out (4 x 1024)
    rowmat_kernel<float><<<dim3(NTOK / 256, B_SZ), 256, 0, stream>>>(
        outl, DMODEL, fc_w, NTOK, fc_b,
        out, NTOK, NTOK, DMODEL, 0);
}
